// KnowledgeInjectionDeepfakeDetector_31550829756930
// MI455X (gfx1250) — matrix operand order repaired and checked by reading, not compiled, not run on hardware
//
#include <hip/hip_runtime.h>
#include <hip/hip_bf16.h>

#define EMBED   768
#define NHEAD   12
#define NPATCH  1024
#define HD      64
#define BATCH   16
#define MTOT    (BATCH * NPATCH)   // 16384

typedef __attribute__((ext_vector_type(16))) __bf16 v16bf;
typedef __attribute__((ext_vector_type(8)))  float  v8f;

union ABReg { v16bf v; unsigned short u[16]; uint4 q[2]; };

__device__ __forceinline__ unsigned short f2bf(float f) {
  unsigned u = __float_as_uint(f);
  u += 0x7FFFu + ((u >> 16) & 1u);          // round-to-nearest-even
  return (unsigned short)(u >> 16);
}

__device__ __forceinline__ v8f wmma_bf16(v16bf a, v16bf b, v8f c) {
  return __builtin_amdgcn_wmma_f32_16x16x32_bf16(false, a, false, b, (short)0, c,
                                                 false, false);
}

// A operand: 16(M) x 32(K) bf16, row-major source with leading dim ld.
// lane&15 = M row; half0 lanes hold K {0..7, 16..23}, half1 {8..15, 24..31}.
__device__ __forceinline__ v16bf load_a16x32(const unsigned short* __restrict__ src,
                                             int ld, int m0, int k0, int lane) {
  const int m  = m0 + (lane & 15);
  const int kk = k0 + ((lane >> 4) << 3);
  const unsigned short* p = src + (size_t)m * ld + kk;
  ABReg r;
  r.q[0] = *reinterpret_cast<const uint4*>(p);
  r.q[1] = *reinterpret_cast<const uint4*>(p + 16);
  return r.v;
}

// B operand: 32(K) x 16(N); source stored column-major: [N rows][K contiguous], ld.
// lane&15 = N col; half0 lanes hold K {0..7, 16..23}, half1 {8..15, 24..31}.
__device__ __forceinline__ v16bf load_b32x16(const unsigned short* __restrict__ src,
                                             int ld, int n0, int k0, int lane) {
  const int n  = n0 + (lane & 15);
  const int kk = k0 + ((lane >> 4) << 3);
  const unsigned short* p = src + (size_t)n * ld + kk;
  ABReg r;
  r.q[0] = *reinterpret_cast<const uint4*>(p);
  r.q[1] = *reinterpret_cast<const uint4*>(p + 16);
  return r.v;
}

// ---- reductions (wave32) -------------------------------------------------
__device__ __forceinline__ float hmax16(float v) {
  v = fmaxf(v, __shfl_xor(v, 1, 32));
  v = fmaxf(v, __shfl_xor(v, 2, 32));
  v = fmaxf(v, __shfl_xor(v, 4, 32));
  v = fmaxf(v, __shfl_xor(v, 8, 32));
  return v;
}
__device__ __forceinline__ float hsum16(float v) {
  v += __shfl_xor(v, 1, 32);
  v += __shfl_xor(v, 2, 32);
  v += __shfl_xor(v, 4, 32);
  v += __shfl_xor(v, 8, 32);
  return v;
}
__device__ __forceinline__ float wsum32(float v) {
#pragma unroll
  for (int m = 16; m >= 1; m >>= 1) v += __shfl_xor(v, m, 32);
  return v;
}
__device__ __forceinline__ float wmax32(float v) {
#pragma unroll
  for (int m = 16; m >= 1; m >>= 1) v = fmaxf(v, __shfl_xor(v, m, 32));
  return v;
}
__device__ __forceinline__ float blockReduceSum(float v, float* sbuf) {
  v = wsum32(v);
  __syncthreads();
  if ((threadIdx.x & 31) == 0) sbuf[threadIdx.x >> 5] = v;
  __syncthreads();
  if (threadIdx.x == 0) {
    float a = sbuf[0];
    for (int i = 1; i < 8; ++i) a += sbuf[i];
    sbuf[0] = a;
  }
  __syncthreads();
  float r = sbuf[0];
  __syncthreads();
  return r;
}
__device__ __forceinline__ float blockReduceMax(float v, float* sbuf) {
  v = wmax32(v);
  __syncthreads();
  if ((threadIdx.x & 31) == 0) sbuf[threadIdx.x >> 5] = v;
  __syncthreads();
  if (threadIdx.x == 0) {
    float a = sbuf[0];
    for (int i = 1; i < 8; ++i) a = fmaxf(a, sbuf[i]);
    sbuf[0] = a;
  }
  __syncthreads();
  float r = sbuf[0];
  __syncthreads();
  return r;
}

// ---- elementwise casts ---------------------------------------------------
__global__ void __launch_bounds__(256)
cast_f32_bf16_kernel(const float* __restrict__ in, unsigned short* __restrict__ out,
                     int n) {
  int i = blockIdx.x * 256 + threadIdx.x;
  if (i < n) out[i] = f2bf(in[i]);
}

// transpose-cast 768x768 weight: out[c*768 + r] = bf16(in[r*768 + c])
__global__ void __launch_bounds__(256)
castT_weight_kernel(const float* __restrict__ in, unsigned short* __restrict__ out) {
  int i = blockIdx.x * 256 + threadIdx.x;  // 589824 total
  int r = i / EMBED, c = i % EMBED;
  out[(size_t)c * EMBED + r] = f2bf(in[i]);
}

// ---- LayerNorm(loc + pos) -> bf16 transposed [B, D, N] -------------------
__global__ void __launch_bounds__(256)
ln_kernel(const float* __restrict__ loc, const float* __restrict__ pos,
          const float* __restrict__ gamma, const float* __restrict__ beta,
          unsigned short* __restrict__ fnT) {
  __shared__ float sbuf[8];
  const int bn = blockIdx.x;
  const int b = bn >> 10, n = bn & 1023;
  const float* x  = loc + (size_t)bn * EMBED;
  const float* pe = pos + (size_t)n * EMBED;
  float ls = 0.f, lq = 0.f;
  for (int d = threadIdx.x; d < EMBED; d += 256) {
    float v = x[d] + pe[d];
    ls += v; lq += v * v;
  }
  float tot  = blockReduceSum(ls, sbuf);
  float totq = blockReduceSum(lq, sbuf);
  float mean = tot * (1.0f / EMBED);
  float var  = totq * (1.0f / EMBED) - mean * mean;
  float rstd = rsqrtf(var + 1e-5f);
  for (int d = threadIdx.x; d < EMBED; d += 256) {
    float v = (x[d] + pe[d] - mean) * rstd * gamma[d] + beta[d];
    fnT[((size_t)b * EMBED + d) * NPATCH + n] = f2bf(v);
  }
}

// ---- generic WMMA GEMM (2x-unrolled ping-pong pipeline) ------------------
// C[M, 768] = A[M, K] (bf16, row-major) x Bt[768, K] (bf16, "col-major" of B)
// Wave computes 2 row-tiles x 4 col-tiles (32x64). Block = 8 waves = 256x64.
// K must be a multiple of 64.
// mode 0: f32 row-major (+bias) | 1: bf16 row-major | 2: bf16 [B,H,N,hd] | 3: bf16 [B,H,hd,N]
__global__ void __launch_bounds__(256)
gemm_bf16_kernel(const unsigned short* __restrict__ A, int lda,
                 const unsigned short* __restrict__ Bt, int ldb,
                 unsigned long long bBatchStride,
                 const float* __restrict__ bias, int K, int mode,
                 float* __restrict__ outF, unsigned short* __restrict__ outU) {
  const int wave = threadIdx.x >> 5, lane = threadIdx.x & 31;
  const int row0 = blockIdx.x * 256 + wave * 32;  // two 16-row tiles
  const int col0 = blockIdx.y * 64;
  const int batch = row0 >> 10;  // rows-per-batch = 1024 (32 | 1024)
  const unsigned short* Bb = Bt + (size_t)batch * bBatchStride;

  v8f acc[2][4];
#pragma unroll
  for (int r = 0; r < 2; ++r)
#pragma unroll
    for (int t = 0; t < 4; ++t)
#pragma unroll
      for (int j = 0; j < 8; ++j) acc[r][t][j] = 0.0f;

  // ping-pong fragment buffers; no rotation copies
  v16bf a0[2], b0[4], a1[2], b1[4];
  a0[0] = load_a16x32(A, lda, row0,      0, lane);
  a0[1] = load_a16x32(A, lda, row0 + 16, 0, lane);
#pragma unroll
  for (int t = 0; t < 4; ++t)
    b0[t] = load_b32x16(Bb, ldb, col0 + t * 16, 0, lane);

  for (int k0 = 0; k0 < K; k0 += 64) {
    // odd-step fragments (k0+32), in flight behind the even-step WMMAs
    a1[0] = load_a16x32(A, lda, row0,      k0 + 32, lane);
    a1[1] = load_a16x32(A, lda, row0 + 16, k0 + 32, lane);
#pragma unroll
    for (int t = 0; t < 4; ++t)
      b1[t] = load_b32x16(Bb, ldb, col0 + t * 16, k0 + 32, lane);

#pragma unroll
    for (int t = 0; t < 4; ++t) {
      acc[0][t] = wmma_bf16(a0[0], b0[t], acc[0][t]);
      acc[1][t] = wmma_bf16(a0[1], b0[t], acc[1][t]);
    }

    const int k2 = k0 + 64;
    if (k2 < K) {  // even-step fragments for next trip
      a0[0] = load_a16x32(A, lda, row0,      k2, lane);
      a0[1] = load_a16x32(A, lda, row0 + 16, k2, lane);
#pragma unroll
      for (int t = 0; t < 4; ++t)
        b0[t] = load_b32x16(Bb, ldb, col0 + t * 16, k2, lane);
    }

#pragma unroll
    for (int t = 0; t < 4; ++t) {
      acc[0][t] = wmma_bf16(a1[0], b1[t], acc[0][t]);
      acc[1][t] = wmma_bf16(a1[1], b1[t], acc[1][t]);
    }
  }

  const int rbase = (lane >> 4) << 3;
  const int cbase = lane & 15;
#pragma unroll
  for (int r = 0; r < 2; ++r) {
    const int rowb = row0 + r * 16 + rbase;  // row of acc element j=0
#pragma unroll
    for (int t = 0; t < 4; ++t) {
      const int col = col0 + t * 16 + cbase;
      const float bv = bias ? bias[col] : 0.0f;
      if (mode == 0) {
        float* p = outF + (size_t)rowb * EMBED + col;
#pragma unroll
        for (int j = 0; j < 8; ++j) p[j * EMBED] = acc[r][t][j] + bv;
      } else if (mode == 1) {
        unsigned short* p = outU + (size_t)rowb * EMBED + col;
#pragma unroll
        for (int j = 0; j < 8; ++j) p[j * EMBED] = f2bf(acc[r][t][j] + bv);
      } else {
        const int h = col >> 6, hd = col & 63;
        const int n0 = rowb & (NPATCH - 1);
        const int b = row0 >> 10;
        if (mode == 2) {
          unsigned short* p =
              outU + (((size_t)(b * NHEAD + h)) * NPATCH + n0) * HD + hd;
#pragma unroll
          for (int j = 0; j < 8; ++j) p[j * HD] = f2bf(acc[r][t][j] + bv);
        } else {
          unsigned short* p =
              outU + (((size_t)(b * NHEAD + h)) * HD + hd) * NPATCH + n0;
#pragma unroll
          for (int j = 0; j < 8; ++j) p[j] = f2bf(acc[r][t][j] + bv);
        }
      }
    }
  }
}

// ---- attention: corr out + online-softmax flash attention ---------------
// grid = B*H*(N/128); block = 256 (8 waves); wave owns 16 rows.
// K/V tiles for the chunk are all issued at the top of the iteration; the
// V waits land after the softmax VALU block and LDS transpose.
__global__ void __launch_bounds__(256)
attn_kernel(const unsigned short* __restrict__ Q, const unsigned short* __restrict__ Kh,
            const unsigned short* __restrict__ Qt, const unsigned short* __restrict__ Vt,
            float* __restrict__ corrOut, float* __restrict__ attnOut) {
  __shared__ __align__(16) unsigned short pbuf[8][512];  // per-wave 16x32 bf16 P tile
  const int wave = threadIdx.x >> 5, lane = threadIdx.x & 31;
  const int rb = blockIdx.x & 7, bh = blockIdx.x >> 3;
  const int b = bh / NHEAD, h = bh % NHEAD;
  const int row0 = rb * 128 + wave * 16;
  const unsigned short* Qb = Q  + (size_t)bh * NPATCH * HD;
  const unsigned short* Tb = Qt + (size_t)bh * NPATCH * HD;
  const unsigned short* Kb = Kh + (size_t)bh * NPATCH * HD;
  const unsigned short* Vb = Vt + (size_t)bh * HD * NPATCH;  // [hd, n]

  const v16bf qa0 = load_a16x32(Qb, HD, row0, 0, lane);
  const v16bf qa1 = load_a16x32(Qb, HD, row0, 32, lane);
  const v16bf ta0 = load_a16x32(Tb, HD, row0, 0, lane);
  const v16bf ta1 = load_a16x32(Tb, HD, row0, 32, lane);

  v8f o[4], mrow, lrow;
#pragma unroll
  for (int j = 0; j < 8; ++j) { mrow[j] = -1e30f; lrow[j] = 0.f; }
#pragma unroll
  for (int d = 0; d < 4; ++d)
#pragma unroll
    for (int j = 0; j < 8; ++j) o[d][j] = 0.f;

  const float inv_scale = 0.125f;           // 1/sqrt(64)
  const int rbase = (lane >> 4) << 3;
  const int cbase = lane & 15;

  // per-wave base pointers with immediate j-strides
  float* crow = corrOut + (size_t)bh * NPATCH * NPATCH +
                (size_t)(row0 + rbase) * NPATCH + cbase;

  for (int mc = 0; mc < NPATCH; mc += 32) {
    // K tiles for this chunk (consumed first), then V tiles (consumed last)
    v16bf kb[2][2];
#pragma unroll
    for (int t = 0; t < 2; ++t) {
      kb[t][0] = load_b32x16(Kb, HD, mc + t * 16, 0, lane);
      kb[t][1] = load_b32x16(Kb, HD, mc + t * 16, 32, lane);
    }
    v16bf vb[4];
#pragma unroll
    for (int d = 0; d < 4; ++d)
      vb[d] = load_b32x16(Vb, NPATCH, d * 16, mc, lane);
    if (mc + 32 < NPATCH)
      __builtin_prefetch(Kb + (size_t)(mc + 32) * HD, 0, 0);

    v8f s[2];
#pragma unroll
    for (int t = 0; t < 2; ++t) {
      v8f cc;
#pragma unroll
      for (int j = 0; j < 8; ++j) cc[j] = 0.f;
      cc = wmma_bf16(ta0, kb[t][0], cc);
      cc = wmma_bf16(ta1, kb[t][1], cc);
#pragma unroll
      for (int j = 0; j < 8; ++j) cc[j] *= inv_scale;
      float* cp = crow + mc + t * 16;
#pragma unroll
      for (int j = 0; j < 8; ++j) cp[j * NPATCH] = cc[j];
      v8f ss;
#pragma unroll
      for (int j = 0; j < 8; ++j) ss[j] = 0.f;
      ss = wmma_bf16(qa0, kb[t][0], ss);
      ss = wmma_bf16(qa1, kb[t][1], ss);
#pragma unroll
      for (int j = 0; j < 8; ++j) ss[j] = ss[j] * inv_scale + cc[j];
      s[t] = ss;
    }

    // online softmax update (row stats live per v8f element; rows split by half)
    v8f mnew, resc, p0, p1;
#pragma unroll
    for (int j = 0; j < 8; ++j) {
      float tm = fmaxf(s[0][j], s[1][j]);
      tm = hmax16(tm);
      mnew[j] = fmaxf(mrow[j], tm);
      resc[j] = __expf(mrow[j] - mnew[j]);
    }
#pragma unroll
    for (int d = 0; d < 4; ++d)
#pragma unroll
      for (int j = 0; j < 8; ++j) o[d][j] *= resc[j];
#pragma unroll
    for (int j = 0; j < 8; ++j) {
      lrow[j] *= resc[j];
      p0[j] = __expf(s[0][j] - mnew[j]);
      p1[j] = __expf(s[1][j] - mnew[j]);
      lrow[j] += hsum16(p0[j] + p1[j]);
      mrow[j] = mnew[j];
    }

    // stage P through LDS: C-layout f32 -> A-operand bf16 (16x32)
    __syncthreads();
#pragma unroll
    for (int j = 0; j < 8; ++j) {
      const int r = j + rbase;
      pbuf[wave][r * 32 + cbase]      = f2bf(p0[j]);
      pbuf[wave][r * 32 + 16 + cbase] = f2bf(p1[j]);
    }
    __syncthreads();
    ABReg pa;
    {
      const int r = lane & 15;
      const int koff = (lane >> 4) << 3;
      pa.q[0] = *reinterpret_cast<const uint4*>(&pbuf[wave][r * 32 + koff]);
      pa.q[1] = *reinterpret_cast<const uint4*>(&pbuf[wave][r * 32 + koff + 16]);
    }
#pragma unroll
    for (int d = 0; d < 4; ++d)
      o[d] = wmma_bf16(pa.v, vb[d], o[d]);
  }

  v8f inv;
#pragma unroll
  for (int j = 0; j < 8; ++j) inv[j] = 1.0f / lrow[j];
  float* op = attnOut + ((size_t)b * NPATCH + row0 + rbase) * EMBED + h * HD + cbase;
#pragma unroll
  for (int d = 0; d < 4; ++d)
#pragma unroll
    for (int j = 0; j < 8; ++j)
      op[j * EMBED + d * 16] = o[d][j] * inv[j];
}

// ---- head-mean of corr + row softmax -> bf16 probs [B, N, N] -------------
__global__ void __launch_bounds__(256)
corr_softmax_kernel(const float* __restrict__ corr, unsigned short* __restrict__ Ploc) {
  __shared__ float rowbuf[NPATCH];
  __shared__ float sbuf[8];
  const int bn = blockIdx.x;
  const int b = bn >> 10, n = bn & 1023;
  const float inv_h = 1.0f / (float)NHEAD;
  float lmax = -1e30f;
  for (int m = threadIdx.x; m < NPATCH; m += 256) {
    float a = 0.f;
#pragma unroll
    for (int h = 0; h < NHEAD; ++h)
      a += corr[(((size_t)(b * NHEAD + h)) * NPATCH + n) * NPATCH + m];
    a *= inv_h;
    rowbuf[m] = a;
    lmax = fmaxf(lmax, a);
  }
  float gmax = blockReduceMax(lmax, sbuf);
  float ls = 0.f;
  for (int m = threadIdx.x; m < NPATCH; m += 256) {
    float e = __expf(rowbuf[m] - gmax);
    rowbuf[m] = e;
    ls += e;
  }
  float gs = blockReduceSum(ls, sbuf);
  float inv = 1.0f / gs;
  for (int m = threadIdx.x; m < NPATCH; m += 256)
    Ploc[(size_t)bn * NPATCH + m] = f2bf(rowbuf[m] * inv);
}

// ---- host ----------------------------------------------------------------
extern "C" void kernel_launch(void* const* d_in, const int* in_sizes, int n_in,
                              void* d_out, int out_size, void* d_ws, size_t ws_size,
                              hipStream_t stream) {
  (void)in_sizes; (void)n_in; (void)out_size; (void)ws_size;

  const float* features = (const float*)d_in[0];
  const float* loc_feat = (const float*)d_in[1];
  const float* w_q      = (const float*)d_in[2];
  const float* b_q      = (const float*)d_in[3];
  const float* w_k      = (const float*)d_in[4];
  const float* b_k      = (const float*)d_in[5];
  const float* w_v      = (const float*)d_in[6];
  const float* b_v      = (const float*)d_in[7];
  const float* w_qt     = (const float*)d_in[8];
  const float* b_qt     = (const float*)d_in[9];
  const float* w_kl     = (const float*)d_in[10];
  const float* b_kl     = (const float*)d_in[11];
  const float* pos_enc  = (const float*)d_in[12];
  const float* ln_gamma = (const float*)d_in[13];
  const float* ln_beta  = (const float*)d_in[14];

  float* outAttn = (float*)d_out;                                  // [16,1024,768]
  float* outLoc  = outAttn + (size_t)MTOT * EMBED;                 // [16,1024,768]
  float* outCorr = outLoc  + (size_t)MTOT * EMBED;                 // [16,12,1024,1024]

  // workspace layout (bf16 elements)
  unsigned short* wsU = (unsigned short*)d_ws;
  const size_t SZ_X  = (size_t)MTOT * EMBED;             // 12,582,912
  const size_t SZ_W  = (size_t)EMBED * EMBED;            // 589,824
  const size_t SZ_P  = (size_t)BATCH * NPATCH * NPATCH;  // 16,777,216
  unsigned short* Xbf   = wsU;
  unsigned short* WqT   = Xbf  + SZ_X;
  unsigned short* WkT   = WqT  + SZ_W;
  unsigned short* WvT   = WkT  + SZ_W;
  unsigned short* WqtT  = WvT  + SZ_W;
  unsigned short* WklT  = WqtT + SZ_W;
  unsigned short* Qh    = WklT + SZ_W;   // [B,H,N,hd]
  unsigned short* Kh    = Qh   + SZ_X;
  unsigned short* Qth   = Kh   + SZ_X;
  unsigned short* Vth   = Qth  + SZ_X;   // [B,H,hd,N]
  unsigned short* fnT   = Vth  + SZ_X;   // [B,D,N]
  unsigned short* Ploc  = fnT  + SZ_X;   // [B,N,N]
  unsigned short* locT  = Ploc + SZ_P;   // [B*N, D] bf16

  // 1) casts
  cast_f32_bf16_kernel<<<(int)((SZ_X + 255) / 256), 256, 0, stream>>>(
      features, Xbf, (int)SZ_X);
  castT_weight_kernel<<<(int)(SZ_W / 256), 256, 0, stream>>>(w_q,  WqT);
  castT_weight_kernel<<<(int)(SZ_W / 256), 256, 0, stream>>>(w_k,  WkT);
  castT_weight_kernel<<<(int)(SZ_W / 256), 256, 0, stream>>>(w_v,  WvT);
  castT_weight_kernel<<<(int)(SZ_W / 256), 256, 0, stream>>>(w_qt, WqtT);
  castT_weight_kernel<<<(int)(SZ_W / 256), 256, 0, stream>>>(w_kl, WklT);

  // 2) LN(loc + pos) -> bf16 transposed
  ln_kernel<<<MTOT, 256, 0, stream>>>(loc_feat, pos_enc, ln_gamma, ln_beta, fnT);

  // 3) projections (WMMA)
  dim3 gGemm(MTOT / 256, EMBED / 64);
  gemm_bf16_kernel<<<gGemm, 256, 0, stream>>>(Xbf, EMBED, WqT,  EMBED, 0ull, b_q,
                                              EMBED, 2, nullptr, Qh);
  gemm_bf16_kernel<<<gGemm, 256, 0, stream>>>(Xbf, EMBED, WkT,  EMBED, 0ull, b_k,
                                              EMBED, 2, nullptr, Kh);
  gemm_bf16_kernel<<<gGemm, 256, 0, stream>>>(Xbf, EMBED, WqtT, EMBED, 0ull, b_qt,
                                              EMBED, 2, nullptr, Qth);
  gemm_bf16_kernel<<<gGemm, 256, 0, stream>>>(Xbf, EMBED, WvT,  EMBED, 0ull, b_v,
                                              EMBED, 3, nullptr, Vth);

  // 4) attention: corr + attn_out (WMMA, online softmax)
  attn_kernel<<<BATCH * NHEAD * (NPATCH / 128), 256, 0, stream>>>(
      Qh, Kh, Qth, Vth, outCorr, outAttn);

  // 5) head-mean + softmax of corr -> bf16 probs
  corr_softmax_kernel<<<MTOT, 256, 0, stream>>>(outCorr, Ploc);

  // 6) localization branch: (Ploc @ fn) @ w_k_loc + b  (WMMA)
  gemm_bf16_kernel<<<gGemm, 256, 0, stream>>>(
      Ploc, NPATCH, fnT, NPATCH, (unsigned long long)EMBED * NPATCH, nullptr,
      NPATCH, 1, nullptr, locT);
  gemm_bf16_kernel<<<gGemm, 256, 0, stream>>>(
      locT, EMBED, WklT, EMBED, 0ull, b_kl, EMBED, 0, outLoc, nullptr);
}
